// GenModel_15616501088320
// MI455X (gfx1250) — matrix operand order repaired and checked by reading, not compiled, not run on hardware
//
#include <hip/hip_runtime.h>
#include <hip/hip_bf16.h>

typedef __attribute__((ext_vector_type(8)))  _Float16 v8h;
typedef __attribute__((ext_vector_type(16))) _Float16 v16h;
typedef __attribute__((ext_vector_type(8)))  float    v8f;

#define NPOINTS 4096
#define KNN     16
#define CH      64

// ---------------------------------------------------------------------------
// Kernel 0: convert w2/w3 (64x64 f32, row-major [out_ch][in_ch]) to f16.
// ---------------------------------------------------------------------------
__global__ __launch_bounds__(256) void cvt_weights(const float* __restrict__ w2,
                                                   const float* __restrict__ w3,
                                                   _Float16* __restrict__ w2h,
                                                   _Float16* __restrict__ w3h) {
  int i = blockIdx.x * 256 + threadIdx.x;
  if (i < CH * CH) {
    w2h[i] = (_Float16)w2[i];
    w3h[i] = (_Float16)w3[i];
  }
}

// ---------------------------------------------------------------------------
// Kernel 1: kNN.  One block per 64 queries; whole batch point cloud in LDS.
// Each wave32 owns 8 queries, one at a time: lanes scan 4096/32 = 128 candidates keeping a
// private sorted top-17 (register arrays, compile-time indices only), then a
// 17-round wave-min merge (shfl_xor, wave32).  Round 0 is the self point
// (dist == 0) and is dropped, matching the reference's [:, :, 1:].
// ---------------------------------------------------------------------------
__global__ __launch_bounds__(256) void knn_kernel(const float* __restrict__ xyz,
                                                  int* __restrict__ knn_out) {
  __shared__ float sp[NPOINTS * 3];            // 48 KB
  const int b    = blockIdx.y;
  const int tid  = threadIdx.x;
  const int lane = tid & 31;
  const int wave = tid >> 5;

  const float* P = xyz + (size_t)b * NPOINTS * 3;
  for (int i = tid; i < NPOINTS * 3; i += 256) sp[i] = P[i];
  __syncthreads();

  for (int qi = 0; qi < 8; ++qi) {
    const int q = blockIdx.x * 64 + wave * 8 + qi;
    const float qx = sp[q * 3 + 0], qy = sp[q * 3 + 1], qz = sp[q * 3 + 2];

    float dist[KNN + 1];
    int   idxs[KNN + 1];
#pragma unroll
    for (int t = 0; t < KNN + 1; ++t) { dist[t] = 3.4e38f; idxs[t] = 0x7fffffff; }

    for (int j = lane; j < NPOINTS; j += 32) {
      const float dx = sp[j * 3 + 0] - qx;
      const float dy = sp[j * 3 + 1] - qy;
      const float dz = sp[j * 3 + 2] - qz;
      const float d  = dx * dx + dy * dy + dz * dz;
      if (d < dist[KNN]) {
        dist[KNN] = d; idxs[KNN] = j;
#pragma unroll
        for (int m = KNN; m > 0; --m) {            // bubble into sorted place
          if (dist[m] < dist[m - 1]) {
            float td = dist[m]; dist[m] = dist[m - 1]; dist[m - 1] = td;
            int   ti = idxs[m]; idxs[m] = idxs[m - 1]; idxs[m - 1] = ti;
          }
        }
      }
    }

    int sel = -1;
#pragma unroll
    for (int t = 0; t < KNN + 1; ++t) {
      float m  = dist[0];
      int   mi = idxs[0];
#pragma unroll
      for (int off = 16; off >= 1; off >>= 1) {     // wave32 min-reduce
        float om  = __shfl_xor(m, off, 32);
        int   omi = __shfl_xor(mi, off, 32);
        if (om < m || (om == m && omi < mi)) { m = om; mi = omi; }
      }
      if (t >= 1 && lane == t - 1) sel = mi;        // lane t-1 keeps neighbor t
      if (dist[0] == m && idxs[0] == mi) {          // winning lane pops its head
#pragma unroll
        for (int s = 0; s < KNN; ++s) { dist[s] = dist[s + 1]; idxs[s] = idxs[s + 1]; }
        dist[KNN] = 3.4e38f; idxs[KNN] = 0x7fffffff;
      }
    }
    if (lane < KNN)
      knn_out[((size_t)b * NPOINTS + q) * KNN + lane] = sel;
  }
}

// ---------------------------------------------------------------------------
// WMMA helper (CDNA5: v_wmma_f32_16x16x32_f16, f32 accumulate).
// ---------------------------------------------------------------------------
__device__ inline v8f wmma32f16(v16h a, v16h b, v8f c) {
  return __builtin_amdgcn_wmma_f32_16x16x32_f16(
      /*neg_a=*/false, a, /*neg_b=*/false, b,
      /*c_mod=*/(short)0, c, /*reuse_a=*/false, /*reuse_b=*/false);
}

// One 64x64 GEMM layer over 16 points (256 rows) of LDS-resident f16
// activations.  Each wave handles 2 points; the 16-row M-tile of each WMMA is
// exactly the 16 neighbors of one point, so FINAL=true pools over k via the
// accumulator VGPRs + one cross-half shfl_xor(16).
template <bool FINAL>
__device__ inline void gemm_layer(const _Float16* __restrict__ src,   // LDS 256x64
                                  const _Float16* __restrict__ w,     // global f16 64x64
                                  _Float16* __restrict__ dst,         // LDS 256x64 (FINAL=false)
                                  float* __restrict__ out,            // (FINAL=true)
                                  int b, int n0, int wave, int lane) {
  const int half = lane >> 4;     // which 16-lane half of the wave
  const int mrow = lane & 15;     // A-fragment row / B-fragment column
#pragma unroll
  for (int pt = 0; pt < 2; ++pt) {
    const int p    = wave * 2 + pt;      // point within block (0..15)
    const int rowb = p * 16;             // 16 neighbor rows of this point
    // A fragments (16x32 f16): lanes 0-15 cover K {0..15}(+k0),
    // lanes 16-31 cover K {16..31}(+k0)
    const _Float16* arow = src + (rowb + mrow) * CH + half * 16;
    v8h  l0 = *(const v8h*)(arow + 0);
    v8h  h0 = *(const v8h*)(arow + 8);
    v16h a0 = __builtin_shufflevector(l0, h0, 0,1,2,3,4,5,6,7,8,9,10,11,12,13,14,15);
    v8h  l1 = *(const v8h*)(arow + 32);
    v8h  h1 = *(const v8h*)(arow + 40);
    v16h a1 = __builtin_shufflevector(l1, h1, 0,1,2,3,4,5,6,7,8,9,10,11,12,13,14,15);
#pragma unroll
    for (int ct = 0; ct < 4; ++ct) {
      // B fragments (32x16 f16): column n = lane&15, K contiguous in the
      // weight row (out = act . w^T, w row-major [out_ch][in_ch]).
      const _Float16* wrow = w + (ct * 16 + mrow) * CH + half * 16;
      v16h b0 = *(const v16h*)(wrow + 0);
      v16h b1 = *(const v16h*)(wrow + 32);
      v8f acc = {};
      acc = wmma32f16(a0, b0, acc);
      acc = wmma32f16(a1, b1, acc);
      if (!FINAL) {
        // D layout: element r -> (M = r + 8*half, N = lane&15)
#pragma unroll
        for (int r = 0; r < 8; ++r)
          dst[(rowb + r + half * 8) * CH + ct * 16 + mrow] =
              (_Float16)fmaxf(acc[r], 0.0f);
      } else {
        float m = acc[0];
#pragma unroll
        for (int r = 1; r < 8; ++r) m = fmaxf(m, acc[r]);
        m = fmaxf(m, __shfl_xor(m, 16, 32));       // combine M halves -> max over all 16 neighbors
        if (lane < 16)
          out[((size_t)b * CH + ct * 16 + lane) * NPOINTS + (n0 + p)] = m;
      }
    }
  }
}

// ---------------------------------------------------------------------------
// Kernel 2: fused gather + rel-coords + (3->64 relu) + 2x WMMA 64x64 layers
// + max-over-k.  16 points (256 rows) per block; dynamic LDS: 2 x 256x64 f16.
// ---------------------------------------------------------------------------
__global__ __launch_bounds__(256) void mlp_kernel(const float* __restrict__ xyz,
                                                  const int* __restrict__ knn,
                                                  const float* __restrict__ w1,
                                                  const _Float16* __restrict__ w2h,
                                                  const _Float16* __restrict__ w3h,
                                                  float* __restrict__ out) {
  extern __shared__ _Float16 smem[];
  _Float16* sA = smem;                 // 256 x 64 activations
  _Float16* sB = smem + 256 * CH;      // 256 x 64 activations

  const int b    = blockIdx.y;
  const int n0   = blockIdx.x * 16;
  const int tid  = threadIdx.x;
  const int lane = tid & 31;
  const int wave = tid >> 5;

  // ---- layer 1: gather neighbor, relative coords, 3->64 FMA + relu ----
  {
    const int r = tid;                 // one row per thread (256 rows)
    const int p = r >> 4, j = r & 15;
    const int q = n0 + p;
    const float* P  = xyz + (size_t)b * NPOINTS * 3;
    const int    nb = knn[((size_t)b * NPOINTS + q) * KNN + j];
    const float rx = P[nb * 3 + 0] - P[q * 3 + 0];
    const float ry = P[nb * 3 + 1] - P[q * 3 + 1];
    const float rz = P[nb * 3 + 2] - P[q * 3 + 2];
#pragma unroll 8
    for (int c = 0; c < CH; ++c) {
      const float h = rx * w1[c * 3 + 0] + ry * w1[c * 3 + 1] + rz * w1[c * 3 + 2];
      sA[r * CH + c] = (_Float16)fmaxf(h, 0.0f);
    }
  }
  __syncthreads();

  // ---- layer 2: 64x64 WMMA + relu -> sB ----
  gemm_layer<false>(sA, w2h, sB, nullptr, b, n0, wave, lane);
  __syncthreads();

  // ---- layer 3: 64x64 WMMA + max over k -> out (B, 64, N) ----
  gemm_layer<true>(sB, w3h, nullptr, out, b, n0, wave, lane);
}

// ---------------------------------------------------------------------------
// Host-side launcher.
// Inputs (setup_inputs order): xyz (B,N,3) f32; w1 (64,3) f32; w2 (64,64) f32;
// w3 (64,64) f32; k (scalar, =16, compile-time here).
// Workspace: [0, 1MB)   kNN indices  (4*4096*16 int32)
//            [1MB, +8K) w2 f16, then w3 f16.
// ---------------------------------------------------------------------------
extern "C" void kernel_launch(void* const* d_in, const int* in_sizes, int n_in,
                              void* d_out, int out_size, void* d_ws, size_t ws_size,
                              hipStream_t stream) {
  const float* xyz = (const float*)d_in[0];
  const float* w1  = (const float*)d_in[1];
  const float* w2  = (const float*)d_in[2];
  const float* w3  = (const float*)d_in[3];
  (void)in_sizes; (void)n_in; (void)out_size; (void)ws_size;

  int*      knn = (int*)d_ws;
  _Float16* w2h = (_Float16*)((char*)d_ws + (1u << 20));
  _Float16* w3h = w2h + CH * CH;
  float*    out = (float*)d_out;

  cvt_weights<<<dim3(16), 256, 0, stream>>>(w2, w3, w2h, w3h);
  knn_kernel<<<dim3(NPOINTS / 64, 4), 256, 0, stream>>>(xyz, knn);
  const size_t lds_bytes = 2u * 256u * CH * sizeof(_Float16);   // 64 KB
  mlp_kernel<<<dim3(NPOINTS / 16, 4), 256, lds_bytes, stream>>>(xyz, knn, w1, w2h, w3h, out);
}
